// MultiHeadAttention_17085379903691
// MI455X (gfx1250) — compile-verified, matrix operand order checked by reading, not run on hardware
//
#include <hip/hip_runtime.h>
#include <hip/hip_bf16.h>

// MI455X / gfx1250: wave32, WMMA 16x16x32 bf16 (f32 accum), TDM tile staging.
// MHA forward: QKV proj (bf16 WMMA GEMM) -> flash attention -> out proj.

typedef __attribute__((ext_vector_type(16))) __bf16 v16bf;
typedef __attribute__((ext_vector_type(8)))  float  v8f;
typedef __attribute__((ext_vector_type(8)))  __bf16 bf16x8;
typedef __attribute__((ext_vector_type(4)))  unsigned int u32x4;
typedef __attribute__((ext_vector_type(8)))  int i32x8;
typedef __attribute__((ext_vector_type(4)))  int i32x4;

static constexpr int BD  = 4;
static constexpr int TT  = 2048;
static constexpr int DD  = 1024;
static constexpr int HH  = 16;
static constexpr int DKK = 64;

#if defined(__has_builtin)
#if __has_builtin(__builtin_amdgcn_tensor_load_to_lds) && \
    __has_builtin(__builtin_amdgcn_s_wait_tensorcnt)
#define USE_TDM 1
#endif
#endif

#define WMMA_BF16(a, b, c) \
  __builtin_amdgcn_wmma_f32_16x16x32_bf16(false, (a), false, (b), (short)0, (c), false, false)

// 16x32 bf16 A/B fragment, contiguous K pairs (ISA 7.12.2):
// VGPR j holds K = kb + 2j (+8 for j>=4). p = base + row(lane)*stride + kb.
__device__ static inline v16bf frag_ld(const __bf16* p) {
  v16bf f;
#pragma unroll
  for (int j = 0; j < 8; ++j) {
    const int k = 2 * j + (j >= 4 ? 8 : 0);
    f[2 * j]     = p[k];
    f[2 * j + 1] = p[k + 1];
  }
  return f;
}

// Same fragment but K runs across LDS rows (row-major tile read as B with
// N = columns): p = base + kb*stride + col(lane).
__device__ static inline v16bf frag_ld_T(const __bf16* p, int stride) {
  v16bf f;
#pragma unroll
  for (int j = 0; j < 8; ++j) {
    const int k = 2 * j + (j >= 4 ? 8 : 0);
    f[2 * j]     = p[k * stride];
    f[2 * j + 1] = p[(k + 1) * stride];
  }
  return f;
}

#if defined(USE_TDM)
// TDM 2-D bf16 tile load: 64 x 32 rows, row = 128B, LDS padded to 144B rows.
// D# fields per CDNA5 ISA.
__device__ static inline void tdm_load_tile(unsigned lds_addr, const __bf16* gsrc) {
  const unsigned long long ga = (unsigned long long)(uintptr_t)gsrc;
  u32x4 g0;
  g0[0] = 1u;                                   // count=1, user descriptor
  g0[1] = lds_addr;                             // lds_addr (bytes)
  g0[2] = (unsigned)ga;                         // global_addr[31:0]
  g0[3] = (unsigned)((ga >> 32) & 0x01FFFFFFull) | (2u << 30);  // addr[56:32], type=2
  i32x8 g1;
  g1[0] = (int)((1u << 16) |       // data_size = 2 bytes
                (1u << 20) |       // pad_enable
                (4u << 22) |       // pad_interval: 32 DWORDs (one 128B row)
                (3u << 25));       // pad_amount: 4 DWORDs (16B -> 144B rows)
  g1[1] = (int)((DKK & 0xFFFFu) << 16);          // tensor_dim0[15:0] = 64
  g1[2] = (int)((32u & 0xFFFFu) << 16);          // tensor_dim0 hi=0 | tensor_dim1=32
  g1[3] = (int)((DKK & 0xFFFFu) << 16);          // tensor_dim1 hi=0 | tile_dim0=64
  g1[4] = (int)(32u);                            // tile_dim1=32 | tile_dim2=0
  g1[5] = (int)DKK;                              // tensor_dim0_stride = 64 elements
  g1[6] = 0;
  g1[7] = 0;
  const i32x4 z4 = {};
#if __clang_major__ >= 23
  const i32x8 z8 = {};
  __builtin_amdgcn_tensor_load_to_lds(g0, g1, z4, z4, z8, 0);
#else
  __builtin_amdgcn_tensor_load_to_lds(g0, g1, z4, z4, 0);
#endif
}
#endif  // USE_TDM

// C = A(8192x1024) * W^T(1024x1024) + bias.
// OUTMODE 0: bf16 into [B][H][T][DK] workspace.  OUTMODE 1: f32 [row][1024].
template <typename AT, int OUTMODE>
__global__ __launch_bounds__(256) void gemm_wtb(const AT* __restrict__ A,
                                                const float* __restrict__ W,
                                                const float* __restrict__ bias,
                                                void* __restrict__ Out) {
  constexpr int Kdim = DD;
  constexpr int ASTR = 40;  // 32 + 8 pad bf16; 16B-aligned, bank-clean
  constexpr int BSTR = 40;
  __shared__ __bf16 sA[64 * ASTR];
  __shared__ __bf16 sB[128 * BSTR];

  const int tid  = threadIdx.x;
  const int wave = tid >> 5;
  const int lane = tid & 31;
  const int wm = wave >> 2, wn = wave & 3;
  const int m0 = blockIdx.x * 64, n0 = blockIdx.y * 128;
  const int lrow  = lane & 15;
  const int kb    = (lane < 16) ? 0 : 8;
  const int rhalf = (lane >= 16) ? 8 : 0;
  const int ar = tid >> 2, ac = (tid & 3) * 8;   // A: 64 x 32
  const int br = tid >> 1, bc = (tid & 1) * 16;  // W: 128 x 32

  auto loadA = [&](int kk, bf16x8& v) {
    const AT* ga = A + (size_t)(m0 + ar) * Kdim + kk + ac;
    if constexpr (sizeof(AT) == 4) {
      const float4 f0 = *(const float4*)(ga);
      const float4 f1 = *(const float4*)(ga + 4);
      v[0] = (__bf16)f0.x; v[1] = (__bf16)f0.y; v[2] = (__bf16)f0.z; v[3] = (__bf16)f0.w;
      v[4] = (__bf16)f1.x; v[5] = (__bf16)f1.y; v[6] = (__bf16)f1.z; v[7] = (__bf16)f1.w;
    } else {
      v = *(const bf16x8*)ga;
    }
  };
  auto loadW = [&](int kk, bf16x8& v0, bf16x8& v1) {
    const float* gw = W + (size_t)(n0 + br) * Kdim + kk + bc;
    const float4 f0 = *(const float4*)(gw);
    const float4 f1 = *(const float4*)(gw + 4);
    const float4 f2 = *(const float4*)(gw + 8);
    const float4 f3 = *(const float4*)(gw + 12);
    v0[0] = (__bf16)f0.x; v0[1] = (__bf16)f0.y; v0[2] = (__bf16)f0.z; v0[3] = (__bf16)f0.w;
    v0[4] = (__bf16)f1.x; v0[5] = (__bf16)f1.y; v0[6] = (__bf16)f1.z; v0[7] = (__bf16)f1.w;
    v1[0] = (__bf16)f2.x; v1[1] = (__bf16)f2.y; v1[2] = (__bf16)f2.z; v1[3] = (__bf16)f2.w;
    v1[4] = (__bf16)f3.x; v1[5] = (__bf16)f3.y; v1[6] = (__bf16)f3.z; v1[7] = (__bf16)f3.w;
  };

  v8f c[2][2] = {};
  bf16x8 ra, rb0, rb1;
  loadA(0, ra);
  loadW(0, rb0, rb1);

  for (int kk = 0; kk < Kdim; kk += 32) {
    __syncthreads();  // previous tile fully consumed
    *(bf16x8*)&sA[ar * ASTR + ac]     = ra;
    *(bf16x8*)&sB[br * BSTR + bc]     = rb0;
    *(bf16x8*)&sB[br * BSTR + bc + 8] = rb1;
    __syncthreads();

    if (kk + 32 < Kdim) {  // pipeline: next slab global->regs while WMMAs run
      loadA(kk + 32, ra);
      loadW(kk + 32, rb0, rb1);
      if (kk + 64 < Kdim) {
        __builtin_prefetch(A + (size_t)(m0 + ar) * Kdim + kk + 64 + ac, 0, 1);
        __builtin_prefetch(W + (size_t)(n0 + br) * Kdim + kk + 64 + bc, 0, 1);
      }
    }

    v16bf af[2], bf[2];
#pragma unroll
    for (int mt = 0; mt < 2; ++mt)
      af[mt] = frag_ld(&sA[(wm * 32 + mt * 16 + lrow) * ASTR + kb]);
#pragma unroll
    for (int nt = 0; nt < 2; ++nt)
      bf[nt] = frag_ld(&sB[(wn * 32 + nt * 16 + lrow) * BSTR + kb]);
#pragma unroll
    for (int mt = 0; mt < 2; ++mt)
#pragma unroll
      for (int nt = 0; nt < 2; ++nt)
        c[mt][nt] = WMMA_BF16(af[mt], bf[nt], c[mt][nt]);
  }

#pragma unroll
  for (int mt = 0; mt < 2; ++mt) {
#pragma unroll
    for (int nt = 0; nt < 2; ++nt) {
      const int col = n0 + wn * 32 + nt * 16 + lrow;
      const float bv = bias[col];
#pragma unroll
      for (int v = 0; v < 8; ++v) {
        const int row = m0 + wm * 32 + mt * 16 + v + rhalf;
        const float val = c[mt][nt][v] + bv;
        if constexpr (OUTMODE == 0) {
          const int bb = row >> 11, t = row & (TT - 1);
          const int h = col >> 6, dk = col & (DKK - 1);
          ((__bf16*)Out)[(((size_t)bb * HH + h) * TT + t) * DKK + dk] = (__bf16)val;
        } else {
          ((float*)Out)[(size_t)row * DD + col] = val;
        }
      }
    }
  }
}

// Flash attention: block = (b, h, 128 q rows); wave owns 16 q rows.
__global__ __launch_bounds__(256) void attn_kernel(const __bf16* __restrict__ Q,
                                                   const __bf16* __restrict__ Kb,
                                                   const __bf16* __restrict__ Vb,
                                                   const unsigned char* __restrict__ mask,
                                                   __bf16* __restrict__ X) {
  constexpr int KSTR = 72;  // 64 + 8 pad bf16 -> 144B rows (TDM pad & 16B align)
  constexpr int PSTR = 36;
  __shared__ __bf16 sK[2][32 * KSTR];     // K tiles: [key][dk], double-buffered
  __shared__ __bf16 sV[2][32 * KSTR];     // V tiles: [key][dk]
  __shared__ __bf16 sP[8][16 * PSTR];     // per-wave P tile: [qrow][key]
  __shared__ unsigned char sMask[TT];

  const int b = blockIdx.z, h = blockIdx.y, qt = blockIdx.x;
  const int tid = threadIdx.x, wave = tid >> 5, lane = tid & 31;
  const int lrow = lane & 15;
  const int kb = (lane < 16) ? 0 : 8;
  const int rhalf = (lane >= 16) ? 8 : 0;

  const size_t headoff = ((size_t)b * HH + h) * TT;
  const int q0 = qt * 128 + wave * 16;

  // stage this batch's mask row once (2KB)
  {
    const unsigned long long mv =
        *(const unsigned long long*)(mask + (size_t)b * TT + tid * 8);
    *(unsigned long long*)&sMask[tid * 8] = mv;
  }

  // Q fragments: 16 rows x 64 (two K=32 fragments), kept in registers
  v16bf qf[2];
  qf[0] = frag_ld(Q + (headoff + q0 + lrow) * DKK + kb);
  qf[1] = frag_ld(Q + (headoff + q0 + lrow) * DKK + kb + 32);

  float m_i[8], l_i[8];
  v8f acc[4] = {};
#pragma unroll
  for (int v = 0; v < 8; ++v) { m_i[v] = -3.0e38f; l_i[v] = 0.f; }

#if !defined(USE_TDM)
  const int skey = tid >> 3;        // fallback staging coords
  const int sdk  = (tid & 7) * 8;
#endif

  constexpr int NIT = TT / 32;
#if defined(USE_TDM)
  if (tid < 32) {  // wave 0 drives the Tensor Data Mover
    tdm_load_tile((unsigned)(uintptr_t)&sK[0][0], Kb + headoff * DKK);
    tdm_load_tile((unsigned)(uintptr_t)&sV[0][0], Vb + headoff * DKK);
  }
#endif
  __syncthreads();  // also covers sMask

  for (int it = 0; it < NIT; ++it) {
    const int cur = it & 1;
    const int kk0 = it * 32;
#if defined(USE_TDM)
    if (tid < 32) {
      if (it + 1 < NIT) {
        tdm_load_tile((unsigned)(uintptr_t)&sK[cur ^ 1][0],
                      Kb + (headoff + kk0 + 32) * DKK);
        tdm_load_tile((unsigned)(uintptr_t)&sV[cur ^ 1][0],
                      Vb + (headoff + kk0 + 32) * DKK);
        __builtin_amdgcn_s_wait_tensorcnt(2);  // in-order: current pair landed
      } else {
        __builtin_amdgcn_s_wait_tensorcnt(0);
      }
    }
    __syncthreads();
#else
    {
      *(bf16x8*)&sK[cur][skey * KSTR + sdk] =
          *(const bf16x8*)(Kb + (headoff + kk0 + skey) * DKK + sdk);
      *(bf16x8*)&sV[cur][skey * KSTR + sdk] =
          *(const bf16x8*)(Vb + (headoff + kk0 + skey) * DKK + sdk);
    }
    __syncthreads();
#endif

    const __bf16* kt = &sK[cur][0];
    const __bf16* vt = &sV[cur][0];

    // S(16x32) = Q(16x64) * K^T, two 16-key column tiles
    v8f s[2];
#pragma unroll
    for (int ns = 0; ns < 2; ++ns) {
      v8f z = {};
      z = WMMA_BF16(qf[0], frag_ld(&kt[(ns * 16 + lrow) * KSTR + kb]), z);
      z = WMMA_BF16(qf[1], frag_ld(&kt[(ns * 16 + lrow) * KSTR + kb + 32]), z);
      s[ns] = z;
    }

    bool mk[2];
#pragma unroll
    for (int ns = 0; ns < 2; ++ns)
      mk[ns] = sMask[kk0 + ns * 16 + lrow] != 0;

    // online softmax per row (row = v + rhalf, values across 16 lanes)
#pragma unroll
    for (int v = 0; v < 8; ++v) {
      const float s0 = mk[0] ? -3.0e38f : s[0][v] * 0.125f;
      const float s1 = mk[1] ? -3.0e38f : s[1][v] * 0.125f;
      float t = fmaxf(s0, s1);
      t = fmaxf(t, __shfl_xor(t, 1, 32));
      t = fmaxf(t, __shfl_xor(t, 2, 32));
      t = fmaxf(t, __shfl_xor(t, 4, 32));
      t = fmaxf(t, __shfl_xor(t, 8, 32));
      const float mnew = fmaxf(m_i[v], t);
      const float al = __expf(m_i[v] - mnew);
      const float p0 = mk[0] ? 0.f : __expf(s0 - mnew);
      const float p1 = mk[1] ? 0.f : __expf(s1 - mnew);
      float ps = p0 + p1;
      ps += __shfl_xor(ps, 1, 32);
      ps += __shfl_xor(ps, 2, 32);
      ps += __shfl_xor(ps, 4, 32);
      ps += __shfl_xor(ps, 8, 32);
      l_i[v] = l_i[v] * al + ps;
      m_i[v] = mnew;
#pragma unroll
      for (int nt = 0; nt < 4; ++nt) acc[nt][v] *= al;
      sP[wave][(v + rhalf) * PSTR + lrow]      = (__bf16)p0;
      sP[wave][(v + rhalf) * PSTR + 16 + lrow] = (__bf16)p1;
    }

    // acc(16x64) += P(16x32) * V(32x64); V read row-major via strided frag
    const v16bf pf = frag_ld(&sP[wave][lrow * PSTR + kb]);
#pragma unroll
    for (int nt = 0; nt < 4; ++nt)
      acc[nt] = WMMA_BF16(pf, frag_ld_T(&vt[kb * KSTR + nt * 16 + lrow], KSTR), acc[nt]);

    __syncthreads();  // all waves done with buf[cur] before TDM/stores reuse it
  }

  // normalize and store X[b][t][h*64+dk] (bf16)
#pragma unroll
  for (int v = 0; v < 8; ++v) {
    const float linv = (l_i[v] > 0.f) ? 1.0f / l_i[v] : 0.f;
    const int trow = q0 + v + rhalf;
#pragma unroll
    for (int nt = 0; nt < 4; ++nt) {
      const int col = h * DKK + nt * 16 + lrow;
      X[((size_t)b * TT + trow) * DD + col] = (__bf16)(acc[nt][v] * linv);
    }
  }
}

extern "C" void kernel_launch(void* const* d_in, const int* in_sizes, int n_in,
                              void* d_out, int out_size, void* d_ws, size_t ws_size,
                              hipStream_t stream) {
  (void)in_sizes; (void)n_in; (void)out_size; (void)ws_size;
  const float* query = (const float*)d_in[0];
  const float* key   = (const float*)d_in[1];
  const float* value = (const float*)d_in[2];
  const unsigned char* mask = (const unsigned char*)d_in[3];
  const float* wq = (const float*)d_in[4];
  const float* bq = (const float*)d_in[5];
  const float* wk = (const float*)d_in[6];
  const float* bk = (const float*)d_in[7];
  const float* wv = (const float*)d_in[8];
  const float* bv = (const float*)d_in[9];
  const float* wo = (const float*)d_in[10];
  const float* bo = (const float*)d_in[11];
  float* out = (float*)d_out;

  const size_t tensElems = (size_t)BD * TT * DD;  // 8M elements
  __bf16* Qb = (__bf16*)d_ws;
  __bf16* Kb = Qb + tensElems;
  __bf16* Vb = Kb + tensElems;
  __bf16* Xb = Vb + tensElems;

  dim3 gGemm(BD * TT / 64, DD / 128);  // 128 x 8
  gemm_wtb<float, 0><<<gGemm, 256, 0, stream>>>(query, wq, bq, (void*)Qb);
  gemm_wtb<float, 0><<<gGemm, 256, 0, stream>>>(key,   wk, bk, (void*)Kb);
  gemm_wtb<float, 0><<<gGemm, 256, 0, stream>>>(value, wv, bv, (void*)Vb);

  dim3 gAttn(TT / 128, HH, BD);
  attn_kernel<<<gAttn, 256, 0, stream>>>(Qb, Kb, Vb, mask, Xb);

  gemm_wtb<__bf16, 1><<<gGemm, 256, 0, stream>>>(Xb, wo, bo, (void*)out);
}
